// AttentionPolyFitScaling_57947698758100
// MI455X (gfx1250) — matrix operand order repaired and checked by reading, not compiled, not run on hardware
//
#include <hip/hip_runtime.h>
#include <hip/hip_bf16.h>

// ---------------------------------------------------------------------------
// AttentionPolyFitScaling on gfx1250 (MI455X): bf16 WMMA GEMMs + two-pass
// streaming (flash-style) attention with adaptive-temperature softmax.
// b=2, n=2048, dim=1024, h=16, hd=64.
// Round 2: branchless online-softmax updates (no EXEC juggling), async
// global->LDS staging of K tiles (ASYNCcnt path), GEMM k-loop unroll.
// ---------------------------------------------------------------------------

typedef __bf16 bf16_t;
typedef __attribute__((ext_vector_type(16))) __bf16 bf16x16;
typedef __attribute__((ext_vector_type(8)))  __bf16 bf16x8;
typedef __attribute__((ext_vector_type(4)))  __bf16 bf16x4;
typedef __attribute__((ext_vector_type(8)))  float  f32x8;

#define SEQ   2048
#define DIM   1024
#define HEADS 16
#define HD    64
#define NEG_MAX (-3.402823466e38f)

static __device__ __forceinline__ f32x8 wmma_bf16(bf16x16 a, bf16x16 b, f32x8 c) {
  // v_wmma_f32_16x16x32_bf16 : D = A(16x32) * B(32x16) + C
  return __builtin_amdgcn_wmma_f32_16x16x32_bf16(false, a, false, b, (short)0, c,
                                                 false, false);
}

static __device__ __forceinline__ bf16x16 cat8(bf16x8 lo, bf16x8 hi) {
  return __builtin_shufflevector(lo, hi, 0, 1, 2, 3, 4, 5, 6, 7,
                                 8, 9, 10, 11, 12, 13, 14, 15);
}

// A-operand (16x32, 16-bit): lane holds M=lane%16, contiguous K16
// [grp*16 .. +16), grp = lane/16.
static __device__ __forceinline__ bf16x16 load_a_frag(const bf16_t* rowPtr,
                                                      int kb, int grp) {
  bf16x8 lo = *(const bf16x8*)(rowPtr + kb + grp * 16);
  bf16x8 hi = *(const bf16x8*)(rowPtr + kb + grp * 16 + 8);
  return cat8(lo, hi);
}

// Async global->LDS copy of 16 bytes per lane (CDNA5 ASYNCcnt path, no VGPR
// round-trip). LDS offset = low 32 bits of the generic pointer.
static __device__ __forceinline__ void async_ld_lds_b128(void* lds,
                                                         const void* gptr) {
  unsigned lds_off = (unsigned)(size_t)lds;
  unsigned long long ga = (unsigned long long)gptr;
  asm volatile("global_load_async_to_lds_b128 %0, %1, off"
               :
               : "v"(lds_off), "v"(ga)
               : "memory");
}

static __device__ __forceinline__ void wait_asynccnt0() {
  asm volatile("s_wait_asynccnt 0x0" ::: "memory");
}

// ---------------------------------------------------------------------------
// fp32 -> bf16 convert (vectorized, grid-stride)
// ---------------------------------------------------------------------------
__global__ void apfs_cvt_bf16(const float* __restrict__ in,
                              bf16_t* __restrict__ out, int n4) {
  int i = blockIdx.x * blockDim.x + threadIdx.x;
  int stride = gridDim.x * blockDim.x;
  for (; i < n4; i += stride) {
    float4 v = ((const float4*)in)[i];
    bf16x4 o;
    o.x = (bf16_t)v.x;
    o.y = (bf16_t)v.y;
    o.z = (bf16_t)v.z;
    o.w = (bf16_t)v.w;
    ((bf16x4*)out)[i] = o;
  }
}

// ---------------------------------------------------------------------------
// GEMM: C = A[4096x1024] * W^T (W rows are K-contiguous => B columns
// contiguous). Block = 256 thr = 8 waves as 2(M) x 4(N); wave tile 32x32.
// MODE 0: store bf16 into per-head QKV layout [(b*16+h), n, hd], scaled.
// MODE 1: store fp32 row-major + bias (final projection).
// ---------------------------------------------------------------------------
template <int MODE>
__global__ __launch_bounds__(256) void apfs_gemm_bf16(
    const bf16_t* __restrict__ A, const bf16_t* __restrict__ W, float scale,
    const float* __restrict__ bias, bf16_t* __restrict__ outb,
    float* __restrict__ outf) {
  constexpr int Kdim = DIM;
  const int lane = threadIdx.x & 31, wave = threadIdx.x >> 5;
  const int grp = lane >> 4, ln = lane & 15;
  const int wm = wave & 1, wn = wave >> 1;
  const int m0 = blockIdx.y * 64 + wm * 32;
  const int n0 = blockIdx.x * 128 + wn * 32;

  const bf16_t* arow0 = A + (size_t)(m0 + ln) * Kdim;
  const bf16_t* arow1 = A + (size_t)(m0 + 16 + ln) * Kdim;
  const bf16_t* wrow0 = W + (size_t)(n0 + ln) * Kdim;
  const bf16_t* wrow1 = W + (size_t)(n0 + 16 + ln) * Kdim;

  f32x8 c[2][2] = {{{}, {}}, {{}, {}}};

#pragma unroll 2
  for (int kb = 0; kb < Kdim; kb += 32) {
    __builtin_prefetch((const void*)(arow0 + kb + 128), 0, 3);  // global_prefetch_b8
    __builtin_prefetch((const void*)(wrow0 + kb + 128), 0, 3);
    bf16x16 a0 = load_a_frag(arow0, kb, grp);
    bf16x16 a1 = load_a_frag(arow1, kb, grp);
    // B-operand: lane N=ln, contiguous K16 at grp*16
    bf16x16 b0 = *(const bf16x16*)(wrow0 + kb + grp * 16);
    bf16x16 b1 = *(const bf16x16*)(wrow1 + kb + grp * 16);
    c[0][0] = wmma_bf16(a0, b0, c[0][0]);
    c[0][1] = wmma_bf16(a0, b1, c[0][1]);
    c[1][0] = wmma_bf16(a1, b0, c[1][0]);
    c[1][1] = wmma_bf16(a1, b1, c[1][1]);
  }

#pragma unroll
  for (int i = 0; i < 2; ++i) {
#pragma unroll
    for (int jn = 0; jn < 2; ++jn) {
#pragma unroll
      for (int r = 0; r < 8; ++r) {
        // C/D layout: row m = r + grp*8, col n = ln
        const int m = m0 + i * 16 + grp * 8 + r;
        const int n = n0 + jn * 16 + ln;
        const float v = c[i][jn][r] * scale;
        if (MODE == 0) {
          const int b = m >> 11, nrow = m & 2047;
          const int hh = n >> 6, d = n & 63;
          outb[((size_t)(b * HEADS + hh) * SEQ + nrow) * HD + d] = (bf16_t)v;
        } else {
          outf[(size_t)m * DIM + n] = v + bias[n];
        }
      }
    }
  }
}

// ---------------------------------------------------------------------------
// Two-pass streaming attention with adaptive-temperature softmax.
// Grid: (16 row-blocks, 32 batch*head). Block = 256 thr = 8 waves; each wave
// owns 16 query rows; K/V tiles (32 x 64) staged cooperatively in LDS
// (K via global_load_async_to_lds_b128).
// Pass 1: online (max, sumexp, sum s*exp) -> entropy -> beta.
// Pass 2: p = exp(beta*s - beta*max), P@V via WMMA (P staged via LDS), /l2.
// All per-element mask handling is branchless (v_cndmask, no EXEC churn).
// ---------------------------------------------------------------------------
__global__ __launch_bounds__(256) void apfs_attn(
    const bf16_t* __restrict__ Qg, const bf16_t* __restrict__ Kg,
    const bf16_t* __restrict__ Vg, bf16_t* __restrict__ Om) {
  __shared__ bf16_t Ksh[32][64];       // K tile, row-major (j, d)
  __shared__ bf16_t Vt[64][32];        // V tile transposed (d, j)
  __shared__ bf16_t Psh[8][16][32];    // per-wave P staging (m, j)

  const int tid = threadIdx.x;
  const int lane = tid & 31, wave = tid >> 5;
  const int grp = lane >> 4, ln = lane & 15;
  const int bh = blockIdx.y;
  const int qblock = blockIdx.x * 128;
  const int m0 = qblock + wave * 16;

  const bf16_t* Qp = Qg + (size_t)bh * SEQ * HD;
  const bf16_t* Kp = Kg + (size_t)bh * SEQ * HD;
  const bf16_t* Vp = Vg + (size_t)bh * SEQ * HD;

  // Q fragments (1/sqrt(hd) pre-folded at projection time), kept in regs.
  const bf16_t* qr = Qp + (size_t)(m0 + ln) * HD;
  const bf16x16 aq0 = load_a_frag(qr, 0, grp);
  const bf16x16 aq1 = load_a_frag(qr, 32, grp);

  const int nJB = qblock / 32 + 4;  // causal: only j <= block max row
  const int cj = tid >> 3;          // coop-load: row within 32-tile
  const int coff = (tid & 7) * 8;   // coop-load: 8-elem column chunk
  bf16_t* kdst = &Ksh[cj][coff];

  float mx[8], lsum[8], tsum[8];
#pragma unroll
  for (int r = 0; r < 8; ++r) { mx[r] = NEG_MAX; lsum[r] = 0.f; tsum[r] = 0.f; }

  // ---------------- Pass 1: entropy statistics ----------------
  for (int jb = 0; jb < nJB; ++jb) {
    __syncthreads();
    async_ld_lds_b128(kdst, Kp + (size_t)(jb * 32 + cj) * HD + coff);
    wait_asynccnt0();
    __syncthreads();
#pragma unroll
    for (int jj = 0; jj < 32; jj += 16) {
      bf16x16 bk0 = *(const bf16x16*)(&Ksh[jj + ln][grp * 16]);
      bf16x16 bk1 = *(const bf16x16*)(&Ksh[jj + ln][32 + grp * 16]);
      f32x8 acc = {};
      acc = wmma_bf16(aq0, bk0, acc);
      acc = wmma_bf16(aq1, bk1, acc);
      const int jg = jb * 32 + jj + ln;
#pragma unroll
      for (int r = 0; r < 8; ++r) {
        const bool ok = (jg <= m0 + grp * 8 + r);
        const float s = acc[r];
        const float nm = ok ? fmaxf(mx[r], s) : mx[r];   // v_cndmask
        const float e = ok ? __expf(s - nm) : 0.0f;      // v_cndmask
        const float w = __expf(mx[r] - nm);
        lsum[r] = fmaf(lsum[r], w, e);
        tsum[r] = fmaf(tsum[r], w, e * s);
        mx[r] = nm;
      }
    }
  }

  // Cross-lane merge (16 lanes of a group hold disjoint j-columns of a row).
#pragma unroll
  for (int off = 8; off >= 1; off >>= 1) {
#pragma unroll
    for (int r = 0; r < 8; ++r) {
      const float om = __shfl_xor(mx[r], off, 16);
      const float ol = __shfl_xor(lsum[r], off, 16);
      const float ot = __shfl_xor(tsum[r], off, 16);
      const float nm = fmaxf(mx[r], om);
      const float w1 = __expf(mx[r] - nm), w2 = __expf(om - nm);
      lsum[r] = lsum[r] * w1 + ol * w2;
      tsum[r] = tsum[r] * w1 + ot * w2;
      mx[r] = nm;
    }
  }

  float beta[8], M2[8], l2[8];
#pragma unroll
  for (int r = 0; r < 8; ++r) {
    const float H = mx[r] + __logf(lsum[r]) - tsum[r] / lsum[r];
    const float poly = fmaf(
        fmaf(fmaf(fmaf(-0.037f, H, 0.481f), H, -2.3f), H, 4.917f), H, -1.791f);
    const float bta = (H > 0.5f) ? fmaxf(poly, 1.0f) : 1.0f;
    beta[r] = bta;
    M2[r] = bta * mx[r];
    l2[r] = 0.0f;
  }

  // ---------------- Pass 2: P = softmax(beta*s), O = P @ V ----------------
  f32x8 o[4] = {{}, {}, {}, {}};
  for (int jb = 0; jb < nJB; ++jb) {
    __syncthreads();
    async_ld_lds_b128(kdst, Kp + (size_t)(jb * 32 + cj) * HD + coff);
    {
      bf16x8 vv = *(const bf16x8*)(Vp + (size_t)(jb * 32 + cj) * HD + coff);
#pragma unroll
      for (int i = 0; i < 8; ++i) Vt[coff + i][cj] = vv[i];  // transpose on write
    }
    wait_asynccnt0();
    __syncthreads();
#pragma unroll
    for (int jj = 0; jj < 32; jj += 16) {
      bf16x16 bk0 = *(const bf16x16*)(&Ksh[jj + ln][grp * 16]);
      bf16x16 bk1 = *(const bf16x16*)(&Ksh[jj + ln][32 + grp * 16]);
      f32x8 acc = {};
      acc = wmma_bf16(aq0, bk0, acc);
      acc = wmma_bf16(aq1, bk1, acc);
      const int jg = jb * 32 + jj + ln;
#pragma unroll
      for (int r = 0; r < 8; ++r) {
        const bool ok = (jg <= m0 + grp * 8 + r);
        const float pe = __expf(beta[r] * acc[r] - M2[r]);
        const float p = ok ? pe : 0.0f;                  // v_cndmask
        l2[r] += p;
        Psh[wave][grp * 8 + r][jj + ln] = (bf16_t)p;     // D-layout -> LDS
      }
    }
    // Re-read P in A-operand layout (wave-private buffer, no barrier needed).
    bf16x8 plo = *(const bf16x8*)(&Psh[wave][ln][grp * 16]);
    bf16x8 phi = *(const bf16x8*)(&Psh[wave][ln][grp * 16 + 8]);
    bf16x16 ap = cat8(plo, phi);
#pragma unroll
    for (int t = 0; t < 4; ++t) {
      bf16x16 bv = *(const bf16x16*)(&Vt[t * 16 + ln][grp * 16]);
      o[t] = wmma_bf16(ap, bv, o[t]);
    }
  }

#pragma unroll
  for (int off = 8; off >= 1; off >>= 1) {
#pragma unroll
    for (int r = 0; r < 8; ++r) l2[r] += __shfl_xor(l2[r], off, 16);
  }

  const int bb = bh >> 4, hh = bh & 15;
#pragma unroll
  for (int t = 0; t < 4; ++t) {
#pragma unroll
    for (int r = 0; r < 8; ++r) {
      const int mrow = m0 + grp * 8 + r;
      const int col = hh * HD + t * 16 + ln;
      Om[(size_t)(bb * SEQ + mrow) * DIM + col] = (bf16_t)(o[t][r] / l2[r]);
    }
  }
}

// ---------------------------------------------------------------------------
// Host launcher
// ---------------------------------------------------------------------------
extern "C" void kernel_launch(void* const* d_in, const int* in_sizes, int n_in,
                              void* d_out, int out_size, void* d_ws,
                              size_t ws_size, hipStream_t stream) {
  (void)in_sizes; (void)n_in; (void)out_size; (void)ws_size;
  const float* x  = (const float*)d_in[0];
  const float* Wq = (const float*)d_in[1];
  const float* Wk = (const float*)d_in[2];
  const float* Wv = (const float*)d_in[3];
  const float* Wo = (const float*)d_in[4];
  const float* bo = (const float*)d_in[5];
  float* out = (float*)d_out;

  const size_t MB = 1024u * 1024u;
  char* w = (char*)d_ws;
  bf16_t* xb  = (bf16_t*)(w);             // [4096,1024] bf16   (8 MB)
  bf16_t* wqb = (bf16_t*)(w + 8 * MB);    // [1024,1024] bf16   (2 MB)
  bf16_t* wkb = (bf16_t*)(w + 10 * MB);
  bf16_t* wvb = (bf16_t*)(w + 12 * MB);
  bf16_t* wob = (bf16_t*)(w + 14 * MB);
  bf16_t* Qb  = (bf16_t*)(w + 16 * MB);   // [32,2048,64] bf16  (8 MB each)
  bf16_t* Kb  = (bf16_t*)(w + 24 * MB);
  bf16_t* Vb  = (bf16_t*)(w + 32 * MB);
  bf16_t* Ob  = (bf16_t*)(w + 40 * MB);   // merged [4096,1024] bf16

  apfs_cvt_bf16<<<2048, 256, 0, stream>>>(x, xb, (2 * SEQ * DIM) / 4);
  apfs_cvt_bf16<<<512, 256, 0, stream>>>(Wq, wqb, (DIM * DIM) / 4);
  apfs_cvt_bf16<<<512, 256, 0, stream>>>(Wk, wkb, (DIM * DIM) / 4);
  apfs_cvt_bf16<<<512, 256, 0, stream>>>(Wv, wvb, (DIM * DIM) / 4);
  apfs_cvt_bf16<<<512, 256, 0, stream>>>(Wo, wob, (DIM * DIM) / 4);

  dim3 ggrid(DIM / 128, (2 * SEQ) / 64);  // (8, 64)
  // Q gets 1/sqrt(hd)=0.125 folded in.
  apfs_gemm_bf16<0><<<ggrid, 256, 0, stream>>>(xb, wqb, 0.125f, nullptr, Qb, nullptr);
  apfs_gemm_bf16<0><<<ggrid, 256, 0, stream>>>(xb, wkb, 1.0f,   nullptr, Kb, nullptr);
  apfs_gemm_bf16<0><<<ggrid, 256, 0, stream>>>(xb, wvb, 1.0f,   nullptr, Vb, nullptr);

  apfs_attn<<<dim3(SEQ / 128, 2 * HEADS), 256, 0, stream>>>(Qb, Kb, Vb, Ob);

  apfs_gemm_bf16<1><<<ggrid, 256, 0, stream>>>(Ob, wob, 1.0f, bo, nullptr, out);
}
